// GAE_25847113187564
// MI455X (gfx1250) — matrix operand order repaired and checked by reading, not compiled, not run on hardware
//
#include <hip/hip_runtime.h>
#include <hip/hip_bf16.h>

typedef __attribute__((ext_vector_type(16))) _Float16 v16h;
typedef __attribute__((ext_vector_type(8)))  _Float16 v8h;
typedef __attribute__((ext_vector_type(4)))  _Float16 v4h;
typedef __attribute__((ext_vector_type(8)))  float    v8f;

#define N_NODES 8192
#define N_EDGES 524288
#define F_IN    128
#define F_HID   256
#define F_OUT   128

// ---------------------------------------------------------------------------
// Small elementwise / setup kernels
// ---------------------------------------------------------------------------

// deg accumulated via float atomics (deg buffer pre-zeroed with memset);
// +1.0 increments are exact in fp32 -> deterministic regardless of order.
__global__ void deg_kernel(const int* __restrict__ dst, float* __restrict__ deg) {
    int e = blockIdx.x * blockDim.x + threadIdx.x;
    atomicAdd(&deg[dst[e]], 1.0f);
}

// dinv = rsqrt(deg + 1)   (+1 = implicit self loop)
__global__ void dinv_kernel(float* __restrict__ deg) {
    int i = blockIdx.x * blockDim.x + threadIdx.x;
    deg[i] = rsqrtf(deg[i] + 1.0f);
}

// gather embedding rows and convert to f16 (4 features per thread)
__global__ void gather_kernel(const int* __restrict__ x,
                              const float* __restrict__ emb,
                              _Float16* __restrict__ h0) {
    int idx = blockIdx.x * blockDim.x + threadIdx.x;   // over N*F_IN/4
    int i = idx >> 5;                                  // /(128/4)
    int f = (idx & 31) << 2;
    const float4 v = *(const float4*)(emb + (size_t)x[i] * F_IN + f);
    v4h o; o[0] = (_Float16)v.x; o[1] = (_Float16)v.y;
           o[2] = (_Float16)v.z; o[3] = (_Float16)v.w;
    *(v4h*)(h0 + (size_t)i * F_IN + f) = o;
}

// transpose + convert weights: Wt[n][k] = (f16) W[k][n]   (W is [K,N] row-major)
__global__ void trans_w_kernel(const float* __restrict__ W,
                               _Float16* __restrict__ Wt, int K, int N) {
    int idx = blockIdx.x * blockDim.x + threadIdx.x;   // over N*K
    int n = idx / K;
    int k = idx % K;
    Wt[idx] = (_Float16)W[(size_t)k * N + n];
}

// ---------------------------------------------------------------------------
// Generic WMMA GEMM:  C[M,N] = A[M,K] * Bt[N,K]^T   (f16 in, f32 out)
// one 16x16 output tile per wave; grids sized exactly (no guards -> EXEC all 1)
// ---------------------------------------------------------------------------
__global__ void gemm_nt_wmma(const _Float16* __restrict__ A,
                             const _Float16* __restrict__ Bt,
                             float* __restrict__ C,
                             int N, int K) {
    int lane = threadIdx.x & 31;
    int wave = threadIdx.x >> 5;
    int tile = blockIdx.x * (blockDim.x >> 5) + wave;
    int ntn  = N >> 4;
    int tm   = tile / ntn;
    int tn   = tile % ntn;

    int m   = (tm << 4) + (lane & 15);
    int n   = (tn << 4) + (lane & 15);
    int kbA = (lane & 16) ? 16 : 0;   // A: lanes 16-31 hold K 16..31
    int kbB = (lane & 16) ? 16 : 0;   // B: lanes 16-31 hold K 16..31

    v8f c = {};
    for (int k0 = 0; k0 < K; k0 += 32) {
        v8h alo = *(const v8h*)(A + (size_t)m * K + k0 + kbA);
        v8h ahi = *(const v8h*)(A + (size_t)m * K + k0 + kbA + 8);
        v16h a;
#pragma unroll
        for (int i = 0; i < 8; ++i) { a[i] = alo[i]; a[i + 8] = ahi[i]; }
        v16h b = *(const v16h*)(Bt + (size_t)n * K + k0 + kbB);
        c = __builtin_amdgcn_wmma_f32_16x16x32_f16(false, a, false, b,
                                                   (short)0, c, false, false);
    }

    int mb = (lane & 16) ? 8 : 0;
#pragma unroll
    for (int r = 0; r < 8; ++r)
        C[(size_t)((tm << 4) + mb + r) * N + n] = c[r];
}

// ---------------------------------------------------------------------------
// Edge aggregation: agg[dst] += hlin[src] * dinv[src]*dinv[dst]
// one thread per (edge, 4-feature chunk); atomics land in L2 (agg is L2-resident)
// ---------------------------------------------------------------------------
__global__ void edge_agg_kernel(const int* __restrict__ src,
                                const int* __restrict__ dst,
                                const float* __restrict__ dinv,
                                const float* __restrict__ hlin,
                                float* __restrict__ agg, int F) {
    int idx = blockIdx.x * blockDim.x + threadIdx.x;   // over E * (F/4)
    int cpe = F >> 2;
    int e   = idx / cpe;
    int c   = (idx - e * cpe) << 2;
    int s = src[e], d = dst[e];
    float norm = dinv[s] * dinv[d];
    const float4 v = *(const float4*)(hlin + (size_t)s * F + c);
    float* a = agg + (size_t)d * F + c;
    atomicAdd(a + 0, v.x * norm);
    atomicAdd(a + 1, v.y * norm);
    atomicAdd(a + 2, v.z * norm);
    atomicAdd(a + 3, v.w * norm);
}

// layer-1 epilogue: h1 = relu(agg + hlin*dinv^2 + b1), convert to f16 (4-wide)
__global__ void post1_kernel(const float* __restrict__ agg,
                             const float* __restrict__ hlin,
                             const float* __restrict__ dinv,
                             const float* __restrict__ b1,
                             _Float16* __restrict__ h1) {
    int idx = blockIdx.x * blockDim.x + threadIdx.x;   // over N*F_HID/4
    int i = idx >> 6;                                  // /(256/4)
    int f = (idx & 63) << 2;
    float di = dinv[i];
    float d2 = di * di;
    size_t base = (size_t)i * F_HID + f;
    const float4 ag = *(const float4*)(agg + base);
    const float4 hl = *(const float4*)(hlin + base);
    const float4 bb = *(const float4*)(b1 + f);
    float r0 = ag.x + hl.x * d2 + bb.x;
    float r1 = ag.y + hl.y * d2 + bb.y;
    float r2 = ag.z + hl.z * d2 + bb.z;
    float r3 = ag.w + hl.w * d2 + bb.w;
    v4h o;
    o[0] = (_Float16)(r0 > 0.0f ? r0 : 0.0f);
    o[1] = (_Float16)(r1 > 0.0f ? r1 : 0.0f);
    o[2] = (_Float16)(r2 > 0.0f ? r2 : 0.0f);
    o[3] = (_Float16)(r3 > 0.0f ? r3 : 0.0f);
    *(v4h*)(h1 + base) = o;
}

// layer-2 epilogue: z = agg + hlin*dinv^2 + b2, convert to f16 (4-wide)
__global__ void post2_kernel(const float* __restrict__ agg,
                             const float* __restrict__ hlin,
                             const float* __restrict__ dinv,
                             const float* __restrict__ b2,
                             _Float16* __restrict__ z) {
    int idx = blockIdx.x * blockDim.x + threadIdx.x;   // over N*F_OUT/4
    int i = idx >> 5;                                  // /(128/4)
    int f = (idx & 31) << 2;
    float di = dinv[i];
    float d2 = di * di;
    size_t base = (size_t)i * F_OUT + f;
    const float4 ag = *(const float4*)(agg + base);
    const float4 hl = *(const float4*)(hlin + base);
    const float4 bb = *(const float4*)(b2 + f);
    v4h o;
    o[0] = (_Float16)(ag.x + hl.x * d2 + bb.x);
    o[1] = (_Float16)(ag.y + hl.y * d2 + bb.y);
    o[2] = (_Float16)(ag.z + hl.z * d2 + bb.z);
    o[3] = (_Float16)(ag.w + hl.w * d2 + bb.w);
    *(v4h*)(z + base) = o;
}

// ---------------------------------------------------------------------------
// Decode: out = sigmoid(Z @ Z^T), Z f16 [8192,128], out f32 [8192,8192]
// Each wave: 16-row strip x 64 cols (4 accum tiles, A fragment reused 4x per
// k-step). Output is a 256MB stream written once -> non-temporal stores so the
// HBM write stream does not thrash the 192MB L2 (Z stays hot in L2/WGP$).
// grid = (8192/64, 8192/128), block = 256 (8 waves)
// ---------------------------------------------------------------------------
__global__ void decode_kernel(const _Float16* __restrict__ Z,
                              float* __restrict__ out) {
    const int K = F_OUT;                 // 128
    int lane = threadIdx.x & 31;
    int wave = threadIdx.x >> 5;
    int m0 = (blockIdx.y * 8 + wave) << 4;
    int n0 = blockIdx.x << 6;            // 4 tiles of 16

    int ml  = m0 + (lane & 15);
    int kbA = (lane & 16) ? 16 : 0;
    int kbB = (lane & 16) ? 16 : 0;

    v8f c0 = {}, c1 = {}, c2 = {}, c3 = {};
    for (int k0 = 0; k0 < K; k0 += 32) {
        v8h alo = *(const v8h*)(Z + (size_t)ml * K + k0 + kbA);
        v8h ahi = *(const v8h*)(Z + (size_t)ml * K + k0 + kbA + 8);
        v16h a;
#pragma unroll
        for (int i = 0; i < 8; ++i) { a[i] = alo[i]; a[i + 8] = ahi[i]; }

        const _Float16* brow = Z + (size_t)(n0 + (lane & 15)) * K + k0 + kbB;
        v16h b0 = *(const v16h*)(brow);
        v16h b1 = *(const v16h*)(brow + 16 * K);
        v16h b2 = *(const v16h*)(brow + 32 * K);
        v16h b3 = *(const v16h*)(brow + 48 * K);
        c0 = __builtin_amdgcn_wmma_f32_16x16x32_f16(false, a, false, b0, (short)0, c0, false, false);
        c1 = __builtin_amdgcn_wmma_f32_16x16x32_f16(false, a, false, b1, (short)0, c1, false, false);
        c2 = __builtin_amdgcn_wmma_f32_16x16x32_f16(false, a, false, b2, (short)0, c2, false, false);
        c3 = __builtin_amdgcn_wmma_f32_16x16x32_f16(false, a, false, b3, (short)0, c3, false, false);
    }

    int mb = (lane & 16) ? 8 : 0;
    int n  = lane & 15;
    v8f acc[4] = {c0, c1, c2, c3};
#pragma unroll
    for (int t = 0; t < 4; ++t) {
#pragma unroll
        for (int r = 0; r < 8; ++r) {
            float v = acc[t][r];
            v = 1.0f / (1.0f + __expf(-v));
            __builtin_nontemporal_store(
                v, out + (size_t)(m0 + mb + r) * N_NODES + n0 + t * 16 + n);
        }
    }
}

// ---------------------------------------------------------------------------
// Host-side launch
// ---------------------------------------------------------------------------
extern "C" void kernel_launch(void* const* d_in, const int* in_sizes, int n_in,
                              void* d_out, int out_size, void* d_ws, size_t ws_size,
                              hipStream_t stream) {
    const int*   x    = (const int*)d_in[0];
    const int*   ei   = (const int*)d_in[1];     // [2,E]: src then dst
    const float* emb  = (const float*)d_in[2];
    const float* W1   = (const float*)d_in[3];   // [128,256]
    const float* b1   = (const float*)d_in[4];
    const float* W2   = (const float*)d_in[5];   // [256,128]
    const float* b2   = (const float*)d_in[6];
    float* out = (float*)d_out;

    const int* src = ei;
    const int* dst = ei + N_EDGES;

    // workspace carve-up (~32MB; all sizes 256B-aligned by construction)
    char* p = (char*)d_ws;
    float*    deg    = (float*)p;                 p += (size_t)N_NODES * 4;               // 32 KB, becomes dinv
    _Float16* h0f16  = (_Float16*)p;              p += (size_t)N_NODES * F_IN * 2;        // 2 MB
    _Float16* W1t    = (_Float16*)p;              p += (size_t)F_HID * F_IN * 2;          // 64 KB
    _Float16* W2t    = (_Float16*)p;              p += (size_t)F_OUT * F_HID * 2;         // 64 KB
    float*    h1lin  = (float*)p;                 p += (size_t)N_NODES * F_HID * 4;       // 8 MB
    float*    agg1   = (float*)p;                 p += (size_t)N_NODES * F_HID * 4;       // 8 MB
    _Float16* h1f16  = (_Float16*)p;              p += (size_t)N_NODES * F_HID * 2;       // 4 MB
    float*    h2lin  = (float*)p;                 p += (size_t)N_NODES * F_OUT * 4;       // 4 MB
    float*    agg2   = (float*)p;                 p += (size_t)N_NODES * F_OUT * 4;       // 4 MB
    _Float16* zf16   = (_Float16*)p;              p += (size_t)N_NODES * F_OUT * 2;       // 2 MB

    // --- degrees / normalization -------------------------------------------
    hipMemsetAsync(deg, 0, (size_t)N_NODES * 4, stream);
    deg_kernel<<<N_EDGES / 256, 256, 0, stream>>>(dst, deg);
    dinv_kernel<<<N_NODES / 256, 256, 0, stream>>>(deg);
    const float* dinv = deg;

    // --- operand prep -------------------------------------------------------
    gather_kernel<<<(N_NODES * F_IN / 4) / 256, 256, 0, stream>>>(x, emb, h0f16);
    trans_w_kernel<<<(F_IN * F_HID) / 256, 256, 0, stream>>>(W1, W1t, F_IN, F_HID);
    trans_w_kernel<<<(F_HID * F_OUT) / 256, 256, 0, stream>>>(W2, W2t, F_HID, F_OUT);

    // --- layer 1: h1lin = h0 @ W1 ; aggregate ; relu -----------------------
    {
        int tiles = (N_NODES / 16) * (F_HID / 16);       // 8192
        gemm_nt_wmma<<<tiles / 8, 256, 0, stream>>>(h0f16, W1t, h1lin, F_HID, F_IN);
    }
    hipMemsetAsync(agg1, 0, (size_t)N_NODES * F_HID * 4, stream);
    edge_agg_kernel<<<(N_EDGES * (F_HID / 4)) / 256, 256, 0, stream>>>(
        src, dst, dinv, h1lin, agg1, F_HID);
    post1_kernel<<<(N_NODES * F_HID / 4) / 256, 256, 0, stream>>>(
        agg1, h1lin, dinv, b1, h1f16);

    // --- layer 2: h2lin = h1 @ W2 ; aggregate ------------------------------
    {
        int tiles = (N_NODES / 16) * (F_OUT / 16);       // 4096
        gemm_nt_wmma<<<tiles / 8, 256, 0, stream>>>(h1f16, W2t, h2lin, F_OUT, F_HID);
    }
    hipMemsetAsync(agg2, 0, (size_t)N_NODES * F_OUT * 4, stream);
    edge_agg_kernel<<<(N_EDGES * (F_OUT / 4)) / 256, 256, 0, stream>>>(
        src, dst, dinv, h2lin, agg2, F_OUT);
    post2_kernel<<<(N_NODES * F_OUT / 4) / 256, 256, 0, stream>>>(
        agg2, h2lin, dinv, b2, zf16);

    // --- decode: sigmoid(z @ z^T) ------------------------------------------
    dim3 grid(N_NODES / 64, N_NODES / 128);
    decode_kernel<<<grid, 256, 0, stream>>>(zf16, out);

    (void)in_sizes; (void)n_in; (void)out_size; (void)ws_size;
}
